// JambaBlock_6923487282557
// MI455X (gfx1250) — matrix operand order repaired and checked by reading, not compiled, not run on hardware
//
#include <hip/hip_runtime.h>
#include <math.h>

// ---------------------------------------------------------------------------
// Types for CDNA5 WMMA (wave32, 16x16x32 bf16 -> f32)
// ---------------------------------------------------------------------------
typedef __bf16 bf16;
typedef __attribute__((ext_vector_type(8)))  __bf16 v8bf;
typedef __attribute__((ext_vector_type(16))) __bf16 v16bf;
typedef __attribute__((ext_vector_type(8)))  float  v8f;

static __device__ __forceinline__ v16bf cat8(v8bf lo, v8bf hi) {
  return __builtin_shufflevector(lo, hi, 0,1,2,3,4,5,6,7,8,9,10,11,12,13,14,15);
}

static __device__ __forceinline__ float silu_f(float v) {
  return v / (1.f + __expf(-v));
}

// ---------------------------------------------------------------------------
// GEMM: C[M,NtilesC*16](f32) = A[M,K](bf16, lda) x W[Np,K](bf16, ldb)^T
//  - W has Np rows, Np a multiple of 128 (zero padded): every wave's two
//    N-tiles are always computable, WMMA never sits under an EXEC mask.
//  - K multiple of 32, M multiple of 128.
//  - A and W K-panels are staged into LDS with gfx1250 async global->LDS
//    copies (ASYNCcnt), double buffered across the K loop.
//  - Per-wave 64x32 output (4 M-tiles x 2 N-tiles); block = 8 waves = 128x128.
// ---------------------------------------------------------------------------
__global__ __launch_bounds__(256) void gemm_bf16_wmma(
    const bf16* __restrict__ A, const bf16* __restrict__ W, float* __restrict__ C,
    int M, int NtilesC, int K, int lda, int ldb, int ldc)
{
  __shared__ __attribute__((aligned(128))) bf16 lA[2][128 * 32];
  __shared__ __attribute__((aligned(128))) bf16 lB[2][128 * 32];

  const int lane  = (int)threadIdx.x;                                   // 0..31
  const int wave  = __builtin_amdgcn_readfirstlane((int)threadIdx.y);   // scalar
  const int tid   = wave * 32 + lane;
  const int waveM = wave >> 2, waveN = wave & 3;
  const int m0    = (int)blockIdx.y * 128;
  const int nb0   = (int)blockIdx.x * 128;   // W row base of this block
  const int n = lane & 15, g = lane >> 4;

  v8f acc[4][2];
#pragma unroll
  for (int i = 0; i < 4; i++)
#pragma unroll
    for (int j = 0; j < 2; j++) acc[i][j] = (v8f){0,0,0,0,0,0,0,0};

  // async stage of one 128x32 bf16 panel: 512 x 16B chunks, 2 per thread/mat.
  auto stageA = [&](int buf, int k0, int c) {
    int row = c >> 2, p = c & 3;
    unsigned long long gp =
        (unsigned long long)(uintptr_t)(A + (size_t)(m0 + row) * lda + k0 + p * 8);
    unsigned lp = (unsigned)(uintptr_t)&lA[buf][row * 32 + p * 8];
    asm volatile("global_load_async_to_lds_b128 %0, %1, off"
                 :: "v"(lp), "v"(gp) : "memory");
  };
  auto stageB = [&](int buf, int k0, int c) {
    int row = c >> 2, p = c & 3;
    unsigned long long gp =
        (unsigned long long)(uintptr_t)(W + (size_t)(nb0 + row) * ldb + k0 + p * 8);
    unsigned lp = (unsigned)(uintptr_t)&lB[buf][row * 32 + p * 8];
    asm volatile("global_load_async_to_lds_b128 %0, %1, off"
                 :: "v"(lp), "v"(gp) : "memory");
  };
  auto stage = [&](int buf, int k0) {
    stageA(buf, k0, tid); stageA(buf, k0, tid + 256);
    stageB(buf, k0, tid); stageB(buf, k0, tid + 256);
  };

  stage(0, 0);
  asm volatile("s_wait_asynccnt 0" ::: "memory");
  __syncthreads();

  int buf = 0;
  for (int k0 = 0; k0 < K; k0 += 32) {
    if (k0 + 32 < K) stage(buf ^ 1, k0 + 32);

    const bf16* lb = &lB[buf][0];
    const bf16* la = &lA[buf][0];
    v16bf b0 = *(const v16bf*)(lb + (waveN * 32 + n) * 32 + g * 16);
    v16bf b1 = *(const v16bf*)(lb + (waveN * 32 + 16 + n) * 32 + g * 16);
#pragma unroll
    for (int i = 0; i < 4; i++) {
      const bf16* pa = la + (waveM * 64 + i * 16 + n) * 32;
      v8bf lo = *(const v8bf*)(pa + g * 16);
      v8bf hi = *(const v8bf*)(pa + 8 + g * 16);
      v16bf a = cat8(lo, hi);
      acc[i][0] = __builtin_amdgcn_wmma_f32_16x16x32_bf16(
          false, a, false, b0, (short)0, acc[i][0], false, false);
      acc[i][1] = __builtin_amdgcn_wmma_f32_16x16x32_bf16(
          false, a, false, b1, (short)0, acc[i][1], false, false);
    }

    asm volatile("s_wait_asynccnt 0" ::: "memory");
    __syncthreads();
    buf ^= 1;
  }

  const int ntbase = __builtin_amdgcn_readfirstlane((int)blockIdx.x * 8 + waveN * 2);
#pragma unroll
  for (int j = 0; j < 2; j++) {
    const int nt = ntbase + j;
    if (nt < NtilesC) {                      // scalar guard, stores only
      const int nc = nt * 16 + n;
#pragma unroll
      for (int i = 0; i < 4; i++) {
#pragma unroll
        for (int r = 0; r < 8; r++) {
          const int mr = m0 + waveM * 64 + i * 16 + r + 8 * g;
          C[(size_t)mr * ldc + nc] = acc[i][j][r];
        }
      }
    }
  }
}

// ---------------------------------------------------------------------------
// Flash attention (causal, GQA). One wave per 16-query tile, full head_dim=256.
// Q bf16 [B*L, H*dh], K bf16 [B*L, KV*dh], Vt bf16 [B,KV,dh,L], O f32 [B*L,H*dh]
// ---------------------------------------------------------------------------
__global__ __launch_bounds__(256) void attn_flash_kernel(
    const bf16* __restrict__ Q, const bf16* __restrict__ Kb,
    const bf16* __restrict__ Vt, float* __restrict__ O,
    int B, int L, int H, int KV, int dh)
{
  __shared__ __attribute__((aligned(32))) bf16 plds[8][16 * 32];
  const int lane = (int)threadIdx.x;
  const int wave = __builtin_amdgcn_readfirstlane((int)threadIdx.y);
  const int qtiles = L / 16;
  const int wid = __builtin_amdgcn_readfirstlane((int)blockIdx.x * 8 + wave);
  const int qt = wid % qtiles;
  const int hh = (wid / qtiles) % H;
  const int b  = wid / (qtiles * H);
  if (b >= B) return;
  const int kv = hh / (H / KV);
  const int n = lane & 15, g = lane >> 4;
  const int ldq = H * dh, ldk = KV * dh;
  const float scale = rsqrtf((float)dh);

  v8f o[16];
#pragma unroll
  for (int t = 0; t < 16; t++) o[t] = (v8f){0,0,0,0,0,0,0,0};
  float mrow[8], lrow[8];
#pragma unroll
  for (int r = 0; r < 8; r++) { mrow[r] = -3.0e38f; lrow[r] = 0.f; }

  const bf16* qbase = Q + (size_t)(b * L + qt * 16 + n) * ldq + hh * dh;
  const bf16* kbase = Kb + (size_t)(b * L) * ldk + kv * dh;
  const bf16* vbase = Vt + (size_t)(b * KV + kv) * dh * (size_t)L;
  bf16* pl = &plds[wave][0];

  const int qhi = qt * 16 + 15;
  for (int j0 = 0; j0 <= qhi; j0 += 32) {
    // ---- S = Q K^T for 32 keys (two 16x16 tiles) ----
    v8f s0 = {0,0,0,0,0,0,0,0}, s1 = {0,0,0,0,0,0,0,0};
#pragma unroll
    for (int k0 = 0; k0 < 256; k0 += 32) {
      v8bf lo = *(const v8bf*)(qbase + k0 + g * 16);
      v8bf hi = *(const v8bf*)(qbase + k0 + 8 + g * 16);
      v16bf aq = cat8(lo, hi);
      v16bf bk0 = *(const v16bf*)(kbase + (size_t)(j0 + n) * ldk + k0 + g * 16);
      v16bf bk1 = *(const v16bf*)(kbase + (size_t)(j0 + 16 + n) * ldk + k0 + g * 16);
      s0 = __builtin_amdgcn_wmma_f32_16x16x32_bf16(false, aq, false, bk0, (short)0, s0, false, false);
      s1 = __builtin_amdgcn_wmma_f32_16x16x32_bf16(false, aq, false, bk1, (short)0, s1, false, false);
    }
    // ---- online softmax ----
    float p0[8], p1[8], sf[8];
#pragma unroll
    for (int r = 0; r < 8; r++) {
      const int row = qt * 16 + r + 8 * g;
      float a0 = (j0 + n      > row) ? -3.0e38f : s0[r] * scale;
      float a1 = (j0 + 16 + n > row) ? -3.0e38f : s1[r] * scale;
      float rmax = fmaxf(a0, a1);
      rmax = fmaxf(rmax, __shfl_xor(rmax, 1, 32));
      rmax = fmaxf(rmax, __shfl_xor(rmax, 2, 32));
      rmax = fmaxf(rmax, __shfl_xor(rmax, 4, 32));
      rmax = fmaxf(rmax, __shfl_xor(rmax, 8, 32));
      float nm = fmaxf(mrow[r], rmax);
      sf[r] = __expf(mrow[r] - nm);
      p0[r] = __expf(a0 - nm);
      p1[r] = __expf(a1 - nm);
      float rs = p0[r] + p1[r];
      rs += __shfl_xor(rs, 1, 32);
      rs += __shfl_xor(rs, 2, 32);
      rs += __shfl_xor(rs, 4, 32);
      rs += __shfl_xor(rs, 8, 32);
      lrow[r] = lrow[r] * sf[r] + rs;
      mrow[r] = nm;
    }
#pragma unroll
    for (int t = 0; t < 16; t++)
#pragma unroll
      for (int r = 0; r < 8; r++) o[t][r] *= sf[r];

    // ---- pack P (16x32) into A-fragment layout via per-wave LDS ----
#pragma unroll
    for (int r = 0; r < 8; r++) {
      pl[(r + 8 * g) * 32 + n]      = (bf16)p0[r];
      pl[(r + 8 * g) * 32 + 16 + n] = (bf16)p1[r];
    }
    asm volatile("s_wait_dscnt 0" ::: "memory");
    v8bf plo = *(const v8bf*)(pl + n * 32 + g * 16);
    v8bf phi = *(const v8bf*)(pl + n * 32 + 8 + g * 16);
    v16bf ap = cat8(plo, phi);

    // ---- O += P V  (V transposed: contiguous keys per dim) ----
#pragma unroll
    for (int t = 0; t < 16; t++) {
      v16bf bv = *(const v16bf*)(vbase + (size_t)(t * 16 + n) * L + j0 + g * 16);
      o[t] = __builtin_amdgcn_wmma_f32_16x16x32_bf16(false, ap, false, bv, (short)0, o[t], false, false);
    }
  }

#pragma unroll
  for (int r = 0; r < 8; r++) {
    const float inv = 1.0f / lrow[r];
    const size_t row = (size_t)(b * L + qt * 16 + r + 8 * g);
#pragma unroll
    for (int t = 0; t < 16; t++)
      O[row * ldq + hh * dh + t * 16 + n] = o[t][r] * inv;
  }
}

// ---------------------------------------------------------------------------
// Elementwise / small kernels (fp32 math)
// ---------------------------------------------------------------------------
__global__ void cvt_pad_kernel(const float* __restrict__ w, bf16* __restrict__ o2,
                               int N, int K, int Np, int Kp)
{
  size_t total = (size_t)Np * Kp;
  size_t i = (size_t)blockIdx.x * blockDim.x + threadIdx.x;
  if (i >= total) return;
  int nn = (int)(i / Kp), kk = (int)(i % Kp);
  float v = (nn < N && kk < K) ? w[(size_t)nn * K + kk] : 0.f;
  o2[i] = (bf16)v;
}

__global__ void cvt_bf16_kernel(const float* __restrict__ in, bf16* __restrict__ o2, size_t total)
{
  size_t i = (size_t)blockIdx.x * blockDim.x + threadIdx.x;
  if (i < total) o2[i] = (bf16)in[i];
}

__global__ void add_kernel(const float* __restrict__ a, const float* __restrict__ b,
                           float* __restrict__ o2, size_t total)
{
  size_t i = (size_t)blockIdx.x * blockDim.x + threadIdx.x;
  if (i < total) o2[i] = a[i] + b[i];
}

__global__ void rmsnorm_bf16_kernel(const float* __restrict__ x, const float* __restrict__ w,
                                    bf16* __restrict__ out, int d)
{
  const size_t row = blockIdx.x;
  const float* xr = x + row * d;
  float ss = 0.f;
  for (int i = threadIdx.x; i < d; i += blockDim.x) { float v = xr[i]; ss += v * v; }
#pragma unroll
  for (int m = 16; m >= 1; m >>= 1) ss += __shfl_xor(ss, m, 32);
  __shared__ float red[8];
  if ((threadIdx.x & 31) == 0) red[threadIdx.x >> 5] = ss;
  __syncthreads();
  float tot = red[0] + red[1] + red[2] + red[3] + red[4] + red[5] + red[6] + red[7];
  float nrm = rsqrtf(tot / (float)d + 1e-6f);
  bf16* orow = out + row * d;
  for (int i = threadIdx.x; i < d; i += blockDim.x)
    orow[i] = (bf16)(xr[i] * nrm * w[i]);
}

// depthwise causal conv (k=4) + SiLU ; reads x_ssm half of xz (ldxz columns)
__global__ void conv_silu_kernel(const float* __restrict__ xz, const float* __restrict__ cw,
                                 const float* __restrict__ cb, float* __restrict__ xs,
                                 bf16* __restrict__ xsbf, int Bb, int L, int D, int ldxz)
{
  size_t total = (size_t)Bb * L * D;
  size_t idx = (size_t)blockIdx.x * blockDim.x + threadIdx.x;
  if (idx >= total) return;
  int d = (int)(idx % D);
  size_t row = idx / D;
  int t = (int)(row % L);
  size_t brow = row - t;
  float acc = cb[d];
#pragma unroll
  for (int k = 0; k < 4; k++) {
    int tt = t + k - 3;
    if (tt >= 0) acc += cw[d * 4 + k] * xz[(brow + tt) * (size_t)ldxz + d];
  }
  float s = silu_f(acc);
  xs[idx] = s;
  xsbf[idx] = (bf16)s;
}

__global__ void dt_softplus_kernel(const float* __restrict__ ssmp, const float* __restrict__ dtw,
                                   const float* __restrict__ dtb, float* __restrict__ delta,
                                   size_t rows, int D, int ldp)
{
  size_t total = rows * D;
  size_t idx = (size_t)blockIdx.x * blockDim.x + threadIdx.x;
  if (idx >= total) return;
  int d = (int)(idx % D);
  size_t row = idx / D;
  float u = ssmp[row * ldp + 32] * dtw[d] + dtb[d];
  delta[idx] = (u > 20.f) ? u : log1pf(__expf(u));
}

__global__ void scan_kernel(const float* __restrict__ xs, const float* __restrict__ delta,
                            const float* __restrict__ ssmp, const float* __restrict__ A_log,
                            const float* __restrict__ Dsk, float* __restrict__ y,
                            int Bb, int L, int D, int ldp)
{
  int idx = blockIdx.x * blockDim.x + threadIdx.x;
  if (idx >= Bb * D) return;
  int b = idx / D, d = idx % D;
  float A[16], h[16];
#pragma unroll
  for (int n2 = 0; n2 < 16; n2++) { A[n2] = -__expf(A_log[d * 16 + n2]); h[n2] = 0.f; }
  float Dv = Dsk[d];
  for (int t = 0; t < L; t++) {
    size_t row = (size_t)b * L + t;
    float xv = xs[row * D + d];
    float dv = delta[row * D + d];
    const float* sp = ssmp + row * ldp;
    float yt = 0.f;
#pragma unroll
    for (int n2 = 0; n2 < 16; n2++) {
      float dA = __expf(dv * A[n2]);
      h[n2] = dA * h[n2] + dv * sp[n2] * xv;
      yt += h[n2] * sp[16 + n2];
    }
    y[row * D + d] = yt + xv * Dv;
  }
}

// y_bf = bf16( y * silu(z) ), z = xz[:, Din:2*Din]
__global__ void gate_mul_kernel(const float* __restrict__ y, const float* __restrict__ xz,
                                bf16* __restrict__ ybf, size_t rows, int D, int ldxz)
{
  size_t total = rows * D;
  size_t idx = (size_t)blockIdx.x * blockDim.x + threadIdx.x;
  if (idx >= total) return;
  int c = (int)(idx % D);
  size_t row = idx / D;
  float z = xz[row * ldxz + D + c];
  ybf[idx] = (bf16)(y[idx] * silu_f(z));
}

// f = bf16( silu(g)*u ) with zero-pad cols >= N (K-pad for down-proj GEMM)
__global__ void silu_mul_kernel(const float* __restrict__ g, const float* __restrict__ u,
                                bf16* __restrict__ f, size_t rows, int N, int ld)
{
  size_t total = rows * (size_t)ld;
  size_t idx = (size_t)blockIdx.x * blockDim.x + threadIdx.x;
  if (idx >= total) return;
  int c = (int)(idx % ld);
  float val = 0.f;
  if (c < N) { float gv = g[idx]; val = silu_f(gv) * u[idx]; }
  f[idx] = (bf16)val;
}

// V [B*L, KV*dh] f32 -> Vt [B,KV,dh,L] bf16 (dim-major for PV B-fragments)
__global__ void v_transpose_kernel(const float* __restrict__ vf, bf16* __restrict__ vt,
                                   int B, int L, int KV, int dh)
{
  size_t total = (size_t)B * L * KV * dh;
  size_t idx = (size_t)blockIdx.x * blockDim.x + threadIdx.x;
  if (idx >= total) return;
  int c = (int)(idx % (KV * dh));
  size_t row = idx / (KV * dh);
  int b = (int)(row / L), t = (int)(row % L);
  int kv = c / dh, d = c % dh;
  vt[(((size_t)(b * KV + kv) * dh + d) * L) + t] = (bf16)vf[idx];
}

// ---------------------------------------------------------------------------
// Host launcher
// ---------------------------------------------------------------------------
extern "C" void kernel_launch(void* const* d_in, const int* in_sizes, int n_in,
                              void* d_out, int out_size, void* d_ws, size_t ws_size,
                              hipStream_t stream)
{
  (void)in_sizes; (void)n_in; (void)out_size; (void)ws_size;
  const float* x      = (const float*)d_in[0];
  const float* norm_m = (const float*)d_in[1];
  const float* w_in   = (const float*)d_in[2];
  const float* conv_w = (const float*)d_in[3];
  const float* conv_b = (const float*)d_in[4];
  const float* w_xp   = (const float*)d_in[5];
  const float* dt_w   = (const float*)d_in[6];
  const float* dt_b   = (const float*)d_in[7];
  const float* A_log  = (const float*)d_in[8];
  const float* D_skip = (const float*)d_in[9];
  const float* w_out  = (const float*)d_in[10];
  const float* norm_a = (const float*)d_in[11];
  const float* w_q    = (const float*)d_in[12];
  const float* w_k    = (const float*)d_in[13];
  const float* w_v    = (const float*)d_in[14];
  const float* w_o    = (const float*)d_in[15];
  const float* norm_f = (const float*)d_in[16];
  const float* w_gate = (const float*)d_in[17];
  const float* w_up   = (const float*)d_in[18];
  const float* w_down = (const float*)d_in[19];
  float* out = (float*)d_out;

  const int Bb = 4, L = 2048, Dm = 1024, Din = 2048, H = 4, KV = 2, dh = 256;
  const int FFN = 2730, FFNk = 2752;       // K-pad32 for down proj
  const int FFNp128 = 2816;                // gate/up N padded to 128
  const int M = Bb * L;                    // 8192
  const size_t Mz = (size_t)M;

  size_t off = 0;
  auto alloc = [&](size_t bytes) -> void* {
    off = (off + 255) & ~(size_t)255;
    void* p = (void*)((char*)d_ws + off);
    off += bytes;
    return p;
  };

  // ---- persistent: bf16 weights (rows padded to 128) + long-lived activations
  bf16* win_bf   = (bf16*)alloc((size_t)2 * Din * 2 * Dm);   // 4096x1024
  bf16* wxp_bf   = (bf16*)alloc((size_t)128 * Din * 2);      // 128x2048 (pad of 33)
  bf16* wout_bf  = (bf16*)alloc((size_t)Dm * Din * 2);
  bf16* wq_bf    = (bf16*)alloc((size_t)Dm * Dm * 2);
  bf16* wk_bf    = (bf16*)alloc((size_t)512 * Dm * 2);
  bf16* wv_bf    = (bf16*)alloc((size_t)512 * Dm * 2);
  bf16* wo_bf    = (bf16*)alloc((size_t)Dm * Dm * 2);
  bf16* wg_bf    = (bf16*)alloc((size_t)FFNp128 * Dm * 2);
  bf16* wu_bf    = (bf16*)alloc((size_t)FFNp128 * Dm * 2);
  bf16* wd_bf    = (bf16*)alloc((size_t)Dm * FFNk * 2);
  bf16* xn_bf    = (bf16*)alloc(Mz * Dm * 2);                // reused by all 3 norms
  float* tmp     = (float*)alloc(Mz * Dm * 4);               // reused GEMM output
  float* hbuf    = (float*)alloc(Mz * Dm * 4);               // mamba output (residual)
  float* x2      = (float*)alloc(Mz * Dm * 4);               // attn output (residual)
  const size_t scr0 = off;                                   // phase-shared scratch base

  const int T = 256;
  auto blocks = [&](size_t tot) { return (unsigned)((tot + T - 1) / T); };
  dim3 gblk(32, 8);
  // Np: padded W rows (mult of 128) -> grid.x ; NtC: 16-col tiles stored to C
  auto gemm = [&](const bf16* A, const bf16* W, float* C, int Np, int NtC, int K,
                  int lda, int ldb, int ldc) {
    dim3 grid(Np / 128, M / 128);
    gemm_bf16_wmma<<<grid, gblk, 0, stream>>>(A, W, C, M, NtC, K, lda, ldb, ldc);
  };
  auto cvtw = [&](const float* w, bf16* o2, int N, int K, int Np, int Kp) {
    size_t tot = (size_t)Np * Kp;
    cvt_pad_kernel<<<blocks(tot), T, 0, stream>>>(w, o2, N, K, Np, Kp);
  };

  // ---- weight conversion (bf16, zero-padded) ----
  cvtw(w_in,   win_bf,  2 * Din, Dm,  2 * Din, Dm);
  cvtw(w_xp,   wxp_bf,  33,      Din, 128,     Din);
  cvtw(w_out,  wout_bf, Dm,      Din, Dm,      Din);
  cvtw(w_q,    wq_bf,   Dm,      Dm,  Dm,      Dm);
  cvtw(w_k,    wk_bf,   512,     Dm,  512,     Dm);
  cvtw(w_v,    wv_bf,   512,     Dm,  512,     Dm);
  cvtw(w_o,    wo_bf,   Dm,      Dm,  Dm,      Dm);
  cvtw(w_gate, wg_bf,   FFN,     Dm,  FFNp128, Dm);
  cvtw(w_up,   wu_bf,   FFN,     Dm,  FFNp128, Dm);
  cvtw(w_down, wd_bf,   Dm,      FFN, Dm,      FFNk);

  // ================= Phase A: Mamba block =================
  off = scr0;
  float* xz    = (float*)alloc(Mz * 4096 * 4);
  float* xs    = (float*)alloc(Mz * Din * 4);
  bf16*  xs_bf = (bf16*)alloc(Mz * Din * 2);
  float* ssmp  = (float*)alloc(Mz * 64 * 4);
  float* delta = (float*)alloc(Mz * Din * 4);
  float* ybuf  = (float*)alloc(Mz * Din * 4);
  bf16*  y_bf  = (bf16*)alloc(Mz * Din * 2);

  rmsnorm_bf16_kernel<<<M, T, 0, stream>>>(x, norm_m, xn_bf, Dm);
  gemm(xn_bf, win_bf, xz, 2 * Din, (2 * Din) / 16, Dm, Dm, Dm, 2 * Din);
  conv_silu_kernel<<<blocks(Mz * Din), T, 0, stream>>>(xz, conv_w, conv_b, xs, xs_bf,
                                                       Bb, L, Din, 2 * Din);
  gemm(xs_bf, wxp_bf, ssmp, 128, 4, Din, Din, Din, 64);
  dt_softplus_kernel<<<blocks(Mz * Din), T, 0, stream>>>(ssmp, dt_w, dt_b, delta, Mz, Din, 64);
  scan_kernel<<<blocks((size_t)Bb * Din), T, 0, stream>>>(xs, delta, ssmp, A_log, D_skip,
                                                          ybuf, Bb, L, Din, 64);
  gate_mul_kernel<<<blocks(Mz * Din), T, 0, stream>>>(ybuf, xz, y_bf, Mz, Din, 2 * Din);
  gemm(y_bf, wout_bf, tmp, Dm, Dm / 16, Din, Din, Din, Dm);
  add_kernel<<<blocks(Mz * Dm), T, 0, stream>>>(x, tmp, hbuf, Mz * Dm);

  // ================= Phase B: attention =================
  off = scr0;
  float* qf   = (float*)alloc(Mz * Dm * 4);
  float* kf   = (float*)alloc(Mz * 512 * 4);
  float* vf   = (float*)alloc(Mz * 512 * 4);
  bf16*  q_bf = (bf16*)alloc(Mz * Dm * 2);
  bf16*  k_bf = (bf16*)alloc(Mz * 512 * 2);
  bf16*  vt_bf= (bf16*)alloc(Mz * 512 * 2);
  float* Obuf = (float*)alloc(Mz * Dm * 4);
  bf16*  O_bf = (bf16*)alloc(Mz * Dm * 2);

  rmsnorm_bf16_kernel<<<M, T, 0, stream>>>(hbuf, norm_a, xn_bf, Dm);
  gemm(xn_bf, wq_bf, qf, Dm, Dm / 16, Dm, Dm, Dm, Dm);
  gemm(xn_bf, wk_bf, kf, 512, 512 / 16, Dm, Dm, Dm, 512);
  gemm(xn_bf, wv_bf, vf, 512, 512 / 16, Dm, Dm, Dm, 512);
  cvt_bf16_kernel<<<blocks(Mz * Dm), T, 0, stream>>>(qf, q_bf, Mz * Dm);
  cvt_bf16_kernel<<<blocks(Mz * 512), T, 0, stream>>>(kf, k_bf, Mz * 512);
  v_transpose_kernel<<<blocks(Mz * 512), T, 0, stream>>>(vf, vt_bf, Bb, L, KV, dh);
  {
    int nwaves = Bb * H * (L / 16);
    attn_flash_kernel<<<nwaves / 8, gblk, 0, stream>>>(q_bf, k_bf, vt_bf, Obuf,
                                                       Bb, L, H, KV, dh);
  }
  cvt_bf16_kernel<<<blocks(Mz * Dm), T, 0, stream>>>(Obuf, O_bf, Mz * Dm);
  gemm(O_bf, wo_bf, tmp, Dm, Dm / 16, Dm, Dm, Dm, Dm);
  add_kernel<<<blocks(Mz * Dm), T, 0, stream>>>(hbuf, tmp, x2, Mz * Dm);

  // ================= Phase C: SwiGLU FFN =================
  off = scr0;
  float* gf  = (float*)alloc(Mz * FFNk * 4);
  float* uf  = (float*)alloc(Mz * FFNk * 4);
  bf16*  f_bf= (bf16*)alloc(Mz * FFNk * 2);

  rmsnorm_bf16_kernel<<<M, T, 0, stream>>>(x2, norm_f, xn_bf, Dm);
  gemm(xn_bf, wg_bf, gf, FFNp128, FFNk / 16, Dm, Dm, Dm, FFNk);
  gemm(xn_bf, wu_bf, uf, FFNp128, FFNk / 16, Dm, Dm, Dm, FFNk);
  silu_mul_kernel<<<blocks(Mz * FFNk), T, 0, stream>>>(gf, uf, f_bf, Mz, FFN, FFNk);
  gemm(f_bf, wd_bf, tmp, Dm, Dm / 16, FFNk, FFNk, FFNk, Dm);
  add_kernel<<<blocks(Mz * Dm), T, 0, stream>>>(x2, tmp, out, Mz * Dm);
}
